// GCN_Drug_Embedder_28157805593353
// MI455X (gfx1250) — matrix operand order repaired and checked by reading, not compiled, not run on hardware
//
#include <hip/hip_runtime.h>
#include <hip/hip_bf16.h>
#include <hip/hip_fp16.h>

typedef __attribute__((ext_vector_type(16))) _Float16 v16h;
typedef __attribute__((ext_vector_type(8)))  float    v8f;
typedef __attribute__((ext_vector_type(4)))  float    f4;

#define N_NODES 16384
#define DIM 64
#define PANEL 256        // K columns of B staged per LDS panel (8 K-steps)

// ---------------------------------------------------------------------------
// K1: h1t[d][n] = relu(embed[idx[n]] @ W1 + b1), f16, TRANSPOSED [64][16384]
// ---------------------------------------------------------------------------
__global__ __launch_bounds__(256) void k1_embed_conv1(
    const int* __restrict__ idx, const float* __restrict__ E,
    const float* __restrict__ W1, const float* __restrict__ b1,
    _Float16* __restrict__ h1t)
{
    __shared__ float w1s[DIM * DIM];
    __shared__ float b1s[DIM];
    __shared__ float es[4][DIM];
    __shared__ float hs[4][DIM];
    const int t = threadIdx.x;
    for (int i = t; i < DIM * DIM; i += 256) w1s[i] = W1[i];
    if (t < DIM) b1s[t] = b1[t];
    const int n0 = blockIdx.x * 4;
    const int g = t >> 6, d = t & 63;
    es[g][d] = E[idx[n0 + g] * DIM + d];
    __syncthreads();
    float acc = b1s[d];
#pragma unroll
    for (int k = 0; k < DIM; ++k) acc = fmaf(es[g][k], w1s[k * DIM + d], acc);
    hs[g][d] = fmaxf(acc, 0.0f);
    __syncthreads();
    h1t[(size_t)(t >> 2) * N_NODES + n0 + (t & 3)] = (_Float16)hs[t & 3][t >> 2];
}

// ---------------------------------------------------------------------------
// K2: p = adj @ h1 via v_wmma_f32_16x16x32_f16, fused with colsum.
// A 16x32 wave tile is all-zero ~36% of the time (density 0.2%): one ballot
// on an OR-reduction skips WMMAs + conversions + colsum entirely. Colsum uses
// pair-gated ballots + popcounts into per-block LDS int counters (exact),
// flushed to global once per 256-column panel.
// ---------------------------------------------------------------------------
__global__ __launch_bounds__(128) void k2_pool1_colsum(
    const float* __restrict__ adj, const _Float16* __restrict__ h1t,
    float* __restrict__ p, float* __restrict__ colsum)
{
    __shared__ _Float16 bb[DIM * PANEL];       // B panel: [d][k], 32 KB
    __shared__ int      csp[PANEL];            // per-panel column counts, 1 KB
    const int t    = threadIdx.x;
    const int lane = t & 31;
    const int w    = t >> 5;
    const int hi   = lane >> 4;                // lane half selects K sub-run
    const int jm   = lane & 15;                // M row (A/D) and N col (B/D)
    const int r0   = blockIdx.x * 64 + w * 16;
    const bool is_flag = (jm == 0);            // lanes 0 and 16 carry counts

    const float* arow = adj + (size_t)(r0 + jm) * N_NODES + hi * 8;
    const int sd = t >> 1;                     // panel staging: d row
    const int sk = (t & 1) * 128;              // panel staging: k half

    v8f c0 = {}, c1 = {}, c2 = {}, c3 = {};

    for (int kp = 0; kp < N_NODES; kp += PANEL) {
        __syncthreads();   // prior panel reads + counter adds complete
        {
            const _Float16* src = h1t + (size_t)sd * N_NODES + kp + sk;
            _Float16* dst = &bb[sd * PANEL + sk];
#pragma unroll
            for (int q = 0; q < 128; q += 8)
                *(f4*)(dst + q) = *(const f4*)(src + q);
        }
        csp[t] = 0;
        csp[t + 128] = 0;
        __syncthreads();   // panel + zeroed counters visible

#pragma unroll 2
        for (int ks = 0; ks < PANEL; ks += 32) {
            const float* ap = arow + kp + ks;
            // A fragment: streamed once -> non-temporal (preserve L2 for h1t/p)
            f4 f0 = __builtin_nontemporal_load((const f4*)ap);
            f4 f1 = __builtin_nontemporal_load((const f4*)(ap + 4));
            f4 f2 = __builtin_nontemporal_load((const f4*)(ap + 16));
            f4 f3 = __builtin_nontemporal_load((const f4*)(ap + 20));
            __builtin_prefetch(ap + 32, 0, 3);      // next K-step's A

            // adj values are bit-exact 0x0 / 0x3f800000 -> integer OR tree
            unsigned xv[16];
            xv[0]  = __float_as_uint(f0.x); xv[1]  = __float_as_uint(f0.y);
            xv[2]  = __float_as_uint(f0.z); xv[3]  = __float_as_uint(f0.w);
            xv[4]  = __float_as_uint(f1.x); xv[5]  = __float_as_uint(f1.y);
            xv[6]  = __float_as_uint(f1.z); xv[7]  = __float_as_uint(f1.w);
            xv[8]  = __float_as_uint(f2.x); xv[9]  = __float_as_uint(f2.y);
            xv[10] = __float_as_uint(f2.z); xv[11] = __float_as_uint(f2.w);
            xv[12] = __float_as_uint(f3.x); xv[13] = __float_as_uint(f3.y);
            xv[14] = __float_as_uint(f3.z); xv[15] = __float_as_uint(f3.w);
            unsigned pr[8];
#pragma unroll
            for (int q = 0; q < 8; ++q) pr[q] = xv[2 * q] | xv[2 * q + 1];
            const unsigned anyv = (pr[0] | pr[1]) | (pr[2] | pr[3]) |
                                  ((pr[4] | pr[5]) | (pr[6] | pr[7]));

            // whole-tile gate: ~36% of K-steps skip everything (tile is zero)
            if (__builtin_amdgcn_ballot_w32(anyv != 0u)) {
                // colsum: pair-gated (each pair nonzero ~12% of the time)
#pragma unroll
                for (int pp = 0; pp < 8; ++pp) {
                    if (__builtin_amdgcn_ballot_w32(pr[pp] != 0u)) {
                        const int j0 = pp * 2;
                        const unsigned m0 = __builtin_amdgcn_ballot_w32(xv[j0] != 0u);
                        const unsigned m1 = __builtin_amdgcn_ballot_w32(xv[j0 + 1] != 0u);
                        const int cnt0 = hi ? __popc(m0 >> 16) : __popc(m0 & 0xFFFFu);
                        const int cnt1 = hi ? __popc(m1 >> 16) : __popc(m1 & 0xFFFFu);
                        const int cb = ks + hi * 8 + ((j0 < 8) ? j0 : j0 + 8);
                        if (is_flag) {                      // exec-mask lanes 0,16
                            atomicAdd(&csp[cb], cnt0);      // ds_add_u32
                            atomicAdd(&csp[cb + 1], cnt1);  // ds_add_u32
                        }
                    }
                }

                v16h a;
                a[0]  = (_Float16)f0.x; a[1]  = (_Float16)f0.y; a[2]  = (_Float16)f0.z; a[3]  = (_Float16)f0.w;
                a[4]  = (_Float16)f1.x; a[5]  = (_Float16)f1.y; a[6]  = (_Float16)f1.z; a[7]  = (_Float16)f1.w;
                a[8]  = (_Float16)f2.x; a[9]  = (_Float16)f2.y; a[10] = (_Float16)f2.z; a[11] = (_Float16)f2.w;
                a[12] = (_Float16)f3.x; a[13] = (_Float16)f3.y; a[14] = (_Float16)f3.z; a[15] = (_Float16)f3.w;

                // B fragments from LDS panel: two 16B runs per lane (k = 8h..8h+7, 16+8h..16+8h+7)
                typedef __attribute__((ext_vector_type(8))) _Float16 v8h_;
                const _Float16* bbase = &bb[jm * PANEL + ks + hi * 8];
                v16h b0 = __builtin_shufflevector(*(const v8h_*)(bbase), *(const v8h_*)(bbase + 16), 0,1,2,3,4,5,6,7,8,9,10,11,12,13,14,15);
                v16h b1 = __builtin_shufflevector(*(const v8h_*)(bbase + 16 * PANEL), *(const v8h_*)(bbase + 16 * PANEL + 16), 0,1,2,3,4,5,6,7,8,9,10,11,12,13,14,15);
                v16h b2 = __builtin_shufflevector(*(const v8h_*)(bbase + 32 * PANEL), *(const v8h_*)(bbase + 32 * PANEL + 16), 0,1,2,3,4,5,6,7,8,9,10,11,12,13,14,15);
                v16h b3 = __builtin_shufflevector(*(const v8h_*)(bbase + 48 * PANEL), *(const v8h_*)(bbase + 48 * PANEL + 16), 0,1,2,3,4,5,6,7,8,9,10,11,12,13,14,15);

                c0 = __builtin_amdgcn_wmma_f32_16x16x32_f16(false, a, false, b0, (short)0, c0, false, false);
                c1 = __builtin_amdgcn_wmma_f32_16x16x32_f16(false, a, false, b1, (short)0, c1, false, false);
                c2 = __builtin_amdgcn_wmma_f32_16x16x32_f16(false, a, false, b2, (short)0, c2, false, false);
                c3 = __builtin_amdgcn_wmma_f32_16x16x32_f16(false, a, false, b3, (short)0, c3, false, false);
            }
        }

        __syncthreads();   // counters complete for this panel
        // flush panel column counts (≈13% nonzero); single gated atomic site
#pragma unroll
        for (int i = 0; i < 2; ++i) {
            const int col = t + i * 128;
            const int cnt = csp[col];
            if (cnt) atomicAdd(&colsum[kp + col], (float)cnt);
        }
    }

    // D layout: lane<16 -> M=v, lane>=16 -> M=8+v; N = jm. Rows owned exclusively.
#pragma unroll
    for (int v = 0; v < 8; ++v) {
        const int row = r0 + hi * 8 + v;
        float* pr2 = p + (size_t)row * DIM + jm;
        pr2[0]  = c0[v];
        pr2[16] = c1[v];
        pr2[32] = c2[v];
        pr2[48] = c3[v];
    }
}

// ---------------------------------------------------------------------------
// K3: out[d] = sum_n colsum[n] * relu(p[n] @ W2 + b2)[d]
// ---------------------------------------------------------------------------
__global__ __launch_bounds__(256) void k3_conv2_readout(
    const float* __restrict__ p, const float* __restrict__ colsum,
    const float* __restrict__ W2, const float* __restrict__ b2,
    float* __restrict__ out)
{
    __shared__ float w2s[DIM * DIM];
    __shared__ float b2s[DIM];
    __shared__ float ps[4][DIM];
    __shared__ float red[256];
    const int t = threadIdx.x;
    for (int i = t; i < DIM * DIM; i += 256) w2s[i] = W2[i];
    if (t < DIM) b2s[t] = b2[t];
    const int g = t >> 6, d = t & 63;
    const int nbase = blockIdx.x * 256;
    float aout = 0.0f;
    for (int it = 0; it < 64; ++it) {
        const int n = nbase + it * 4 + g;
        __syncthreads();
        ps[g][d] = p[(size_t)n * DIM + d];
        __syncthreads();
        const float c = colsum[n];
        float acc = b2s[d];
#pragma unroll
        for (int k = 0; k < DIM; ++k) acc = fmaf(ps[g][k], w2s[k * DIM + d], acc);
        aout = fmaf(c, fmaxf(acc, 0.0f), aout);
    }
    red[t] = aout;
    __syncthreads();
    if (t < 64) {
        const float s = red[t] + red[t + 64] + red[t + 128] + red[t + 192];
        atomicAdd(&out[t], s);
    }
}

// ---------------------------------------------------------------------------
extern "C" void kernel_launch(void* const* d_in, const int* in_sizes, int n_in,
                              void* d_out, int out_size, void* d_ws, size_t ws_size,
                              hipStream_t stream) {
    const int*   idx = (const int*)d_in[0];
    const float* adj = (const float*)d_in[1];
    const float* E   = (const float*)d_in[2];
    const float* W1  = (const float*)d_in[3];
    const float* b1  = (const float*)d_in[4];
    const float* W2  = (const float*)d_in[5];
    const float* b2  = (const float*)d_in[6];
    float* out = (float*)d_out;

    char* ws = (char*)d_ws;
    float*    colsum = (float*)ws;                              // 64 KB
    _Float16* h1t    = (_Float16*)(ws + 65536);                 // 2 MB
    float*    p      = (float*)(ws + 65536 + 2097152);          // 4 MB

    hipMemsetAsync(colsum, 0, N_NODES * sizeof(float), stream);
    hipMemsetAsync(out, 0, DIM * sizeof(float), stream);

    k1_embed_conv1<<<N_NODES / 4, 256, 0, stream>>>(idx, E, W1, b1, h1t);
    k2_pool1_colsum<<<N_NODES / 64, 128, 0, stream>>>(adj, h1t, p, colsum);
    k3_conv2_readout<<<N_NODES / 256, 256, 0, stream>>>(p, colsum, W2, b2, out);
}
